// DecoderAMRPALayer_5093831213316
// MI455X (gfx1250) — compile-verified, matrix operand order checked by reading, not run on hardware
//
#include <hip/hip_runtime.h>
#include <math.h>

// ---------- types ----------
typedef unsigned short bf16s; // raw bf16 storage (bit pattern)
typedef __attribute__((ext_vector_type(16))) __bf16 v16bf;
typedef __attribute__((ext_vector_type(8)))  __bf16 v8bf;
typedef __attribute__((ext_vector_type(8)))  float  v8f;
typedef unsigned int u32x4 __attribute__((ext_vector_type(4)));
typedef int          i32x4 __attribute__((ext_vector_type(4)));
typedef int          i32x8 __attribute__((ext_vector_type(8)));

union FragAB { v16bf v; v8bf h[2]; };

#if defined(__HIP_DEVICE_COMPILE__) && __has_builtin(__builtin_amdgcn_tensor_load_to_lds)
#define HAVE_TDM 1
#else
#define HAVE_TDM 0
#endif

__device__ __forceinline__ bf16s f2b(float f) {
  union { float f; unsigned u; } v; v.f = f;
  unsigned r = (v.u + 0x7FFFu + ((v.u >> 16) & 1u)) >> 16; // RNE
  return (bf16s)r;
}
__device__ __forceinline__ float b2f(bf16s b) {
  union { unsigned u; float f; } v; v.u = ((unsigned)b) << 16;
  return v.f;
}

template <typename T> __device__ __forceinline__ T cvt_out(float f);
template <> __device__ __forceinline__ float cvt_out<float>(float f) { return f; }
template <> __device__ __forceinline__ bf16s cvt_out<bf16s>(float f) { return f2b(f); }

__device__ __forceinline__ unsigned lds_off32(const void* p) {
  return (unsigned)(unsigned long long)p; // generic->LDS offset: low 32 bits
}

// ---------- TDM 2-D tile load: global (row-major, stride in elements) -> LDS ----------
// Tile: tile_d1 rows x 32 bf16 elements; LDS layout adds 16B pad after every 64B row
// (pad_interval code 3 = 16 DWORDs, pad_amount code 3 = 4 DWORDs) to match LDS_STRIDE=40.
__device__ __forceinline__ void tdm_load_tile(unsigned lds_addr, const bf16s* gptr,
                                              unsigned tile_d1,
                                              unsigned long long stride_elems) {
#if HAVE_TDM
  unsigned long long ga = (unsigned long long)gptr;
  u32x4 g0;
  g0[0] = 1u;                                   // count=1, user descriptor
  g0[1] = lds_addr;                             // lds_addr (bytes)
  g0[2] = (unsigned)(ga & 0xFFFFFFFFu);         // global_addr[31:0]
  g0[3] = (unsigned)((ga >> 32) & 0x1FFFFFFu) | (2u << 30); // addr[56:32], type=2
  i32x8 g1;
  // word0: wg_mask=0 | data_size=1(2B) | pad_enable | pad_interval=3 | pad_amount=3
  g1[0] = (int)((1u << 16) | (1u << 20) | (3u << 22) | (3u << 25));
  const unsigned td0 = 0x7FFFFFFFu;             // tensor_dim0: huge, no OOB clip
  const unsigned td1 = 0x7FFFFFFFu;             // tensor_dim1: huge, no OOB clip
  g1[1] = (int)((td0 & 0xFFFFu) << 16);         // [63:48]  = dim0 lo16
  g1[2] = (int)(((td0 >> 16) & 0xFFFFu) | ((td1 & 0xFFFFu) << 16)); // dim0 hi16 | dim1 lo16
  g1[3] = (int)(((td1 >> 16) & 0xFFFFu) | (32u << 16));             // dim1 hi16 | tile_dim0=32
  g1[4] = (int)(tile_d1 & 0xFFFFu);             // tile_dim1; tile_dim2=0
  g1[5] = (int)(stride_elems & 0xFFFFFFFFu);    // tensor_dim0_stride lo32
  g1[6] = (int)((stride_elems >> 32) & 0xFFFFu);// tensor_dim0_stride hi16
  g1[7] = 0;
  i32x4 z4 = {0, 0, 0, 0};
#if defined(__clang_major__) && __clang_major__ >= 23
  i32x8 z8 = {0, 0, 0, 0, 0, 0, 0, 0};
  __builtin_amdgcn_tensor_load_to_lds(g0, g1, z4, z4, z8, 0);
#else
  __builtin_amdgcn_tensor_load_to_lds(g0, g1, z4, z4, 0);
#endif
#else
  (void)lds_addr; (void)gptr; (void)tile_d1; (void)stride_elems;
#endif
}

// ---------- WMMA GEMM: C[M,N] = alpha * A[M,K] * B[N,K]^T + bias ----------
// A: row-major [M, lda], bf16. B: row-major [N, ldb] (i.e. N x K), bf16.
// 256 threads = 8 waves; block tile 128x64, K tile 32; wave tile 32x32 (2x2 WMMA).
#define BM 128
#define BN 64
#define BK 32
#define LDS_STRIDE 40  // 32 + 8 pad, keeps 16B alignment (64B data + 16B pad per row)

template <typename TOut>
__global__ __launch_bounds__(256) void wmma_gemm_nt(
    const bf16s* __restrict__ A, long long lda, long long strideA,
    const bf16s* __restrict__ B, long long ldb, long long strideB,
    TOut* __restrict__ C, long long ldc, long long strideC,
    int M, int N, int K, float alpha, const float* __restrict__ bias)
{
  const int bz = blockIdx.z;
  A += (size_t)bz * strideA;
  B += (size_t)bz * strideB;
  C += (size_t)bz * strideC;

  const int m0 = blockIdx.y * BM;
  const int n0 = blockIdx.x * BN;
  const int tid  = threadIdx.x;
  const int wid  = tid >> 5;
  const int lane = tid & 31;
  const int wm   = wid >> 1;       // 0..3  -> wave row (32 rows each)
  const int wn   = wid & 1;        // 0..1  -> wave col (32 cols each)
  const int half = lane >> 4;      // ISA: lanes 16-31 hold the K+8 slice
  const int lrow = lane & 15;

  v8f acc[2][2];
  for (int i = 0; i < 2; ++i)
    for (int j = 0; j < 2; ++j)
      for (int e = 0; e < 8; ++e) acc[i][j][e] = 0.0f;

#if HAVE_TDM
  // ---- TDM double-buffered pipeline: wave 0 drives the Tensor Data Mover ----
  __shared__ __align__(16) bf16s As[2][BM * LDS_STRIDE];
  __shared__ __align__(16) bf16s Bs[2][BN * LDS_STRIDE];
  const int nt = K / BK;
  if (wid == 0) {
    tdm_load_tile(lds_off32(&As[0][0]), A + (size_t)m0 * lda, BM, (unsigned long long)lda);
    tdm_load_tile(lds_off32(&Bs[0][0]), B + (size_t)n0 * ldb, BN, (unsigned long long)ldb);
  }
  for (int t = 0; t < nt; ++t) {
    const int cur = t & 1;
    if (wid == 0) __builtin_amdgcn_s_wait_tensorcnt((short)0); // tile t landed in LDS
    __syncthreads();
    if (wid == 0 && (t + 1) < nt) { // prefetch tile t+1 into the other buffer
      const int nxt = cur ^ 1;
      const long long k1 = (long long)(t + 1) * BK;
      tdm_load_tile(lds_off32(&As[nxt][0]), A + (size_t)m0 * lda + k1, BM, (unsigned long long)lda);
      tdm_load_tile(lds_off32(&Bs[nxt][0]), B + (size_t)n0 * ldb + k1, BN, (unsigned long long)ldb);
    }
    FragAB fa[2], fb[2];
    for (int i = 0; i < 2; ++i) {
      const v8bf* p = reinterpret_cast<const v8bf*>(
          &As[cur][(wm * 32 + i * 16 + lrow) * LDS_STRIDE + half * 8]);
      fa[i].h[0] = p[0];   // K = half*8 .. half*8+7
      fa[i].h[1] = p[2];   // K = half*8+16 .. half*8+23
    }
    for (int j = 0; j < 2; ++j) {
      const v8bf* p = reinterpret_cast<const v8bf*>(
          &Bs[cur][(wn * 32 + j * 16 + lrow) * LDS_STRIDE + half * 8]);
      fb[j].h[0] = p[0];
      fb[j].h[1] = p[2];
    }
    for (int i = 0; i < 2; ++i)
      for (int j = 0; j < 2; ++j)
        acc[i][j] = __builtin_amdgcn_wmma_f32_16x16x32_bf16(
            false, fa[i].v, false, fb[j].v, (short)0, acc[i][j], false, false);
    // no trailing barrier needed: each wave's LDS reads complete (s_wait_dscnt
    // precedes its WMMA and barrier arrival) before wave 0 can issue the next
    // TDM overwrite of this buffer after the next top-of-loop barrier.
  }
#else
  // ---- fallback: per-thread global->LDS staging ----
  __shared__ __align__(16) bf16s As[1][BM * LDS_STRIDE];
  __shared__ __align__(16) bf16s Bs[1][BN * LDS_STRIDE];
  const int arow = tid >> 1, aseg = tid & 1;  // A: 128 rows x 2 segments of 16 elems
  const int brow = tid >> 2, bseg = tid & 3;  // B:  64 rows x 4 segments of 8 elems
  for (int k0 = 0; k0 < K; k0 += BK) {
    {
      const uint4* g = reinterpret_cast<const uint4*>(A + (size_t)(m0 + arow) * lda + k0 + aseg * 16);
      uint4* s = reinterpret_cast<uint4*>(&As[0][arow * LDS_STRIDE + aseg * 16]);
      s[0] = g[0];
      s[1] = g[1];
    }
    {
      const uint4* g = reinterpret_cast<const uint4*>(B + (size_t)(n0 + brow) * ldb + k0 + bseg * 8);
      uint4* s = reinterpret_cast<uint4*>(&Bs[0][brow * LDS_STRIDE + bseg * 8]);
      s[0] = g[0];
    }
    if (k0 + BK < K) {
      __builtin_prefetch(A + (size_t)(m0 + arow) * lda + k0 + BK + aseg * 16, 0, 1);
      __builtin_prefetch(B + (size_t)(n0 + brow) * ldb + k0 + BK + bseg * 8, 0, 1);
    }
    __syncthreads();
    FragAB fa[2], fb[2];
    for (int i = 0; i < 2; ++i) {
      const v8bf* p = reinterpret_cast<const v8bf*>(
          &As[0][(wm * 32 + i * 16 + lrow) * LDS_STRIDE + half * 8]);
      fa[i].h[0] = p[0];
      fa[i].h[1] = p[2];
    }
    for (int j = 0; j < 2; ++j) {
      const v8bf* p = reinterpret_cast<const v8bf*>(
          &Bs[0][(wn * 32 + j * 16 + lrow) * LDS_STRIDE + half * 8]);
      fb[j].h[0] = p[0];
      fb[j].h[1] = p[2];
    }
    for (int i = 0; i < 2; ++i)
      for (int j = 0; j < 2; ++j)
        acc[i][j] = __builtin_amdgcn_wmma_f32_16x16x32_bf16(
            false, fa[i].v, false, fb[j].v, (short)0, acc[i][j], false, false);
    __syncthreads();
  }
#endif

  // store: D layout -> VGPR r : M = half*8 + r, N = lrow
  for (int i = 0; i < 2; ++i) {
    const int gmb = m0 + wm * 32 + i * 16 + half * 8;
    for (int j = 0; j < 2; ++j) {
      const int gn = n0 + wn * 32 + j * 16 + lrow;
      const float bv = bias ? bias[gn] : 0.0f;
      for (int r = 0; r < 8; ++r)
        C[(size_t)(gmb + r) * ldc + gn] = cvt_out<TOut>(acc[i][j][r] * alpha + bv);
    }
  }
}

// ---------- elementwise / transforms ----------
__global__ __launch_bounds__(256) void f32_to_bf16_kernel(const float* __restrict__ in,
                                                          bf16s* __restrict__ out, size_t n) {
  for (size_t i = (size_t)blockIdx.x * blockDim.x + threadIdx.x; i < n;
       i += (size_t)gridDim.x * blockDim.x)
    out[i] = f2b(in[i]);
}

__global__ __launch_bounds__(256) void transpose_f32_to_bf16_kernel(
    const float* __restrict__ in, long long ldin, bf16s* __restrict__ out, long long ldout) {
  __shared__ float tile[32][33];
  const int c0 = blockIdx.x * 32, r0 = blockIdx.y * 32;
  const int tx = threadIdx.x & 31, ty = threadIdx.x >> 5;
  for (int rr = ty; rr < 32; rr += 8)
    tile[rr][tx] = in[(size_t)(r0 + rr) * ldin + c0 + tx];
  __syncthreads();
  for (int rr = ty; rr < 32; rr += 8)
    out[(size_t)(c0 + rr) * ldout + r0 + tx] = f2b(tile[tx][rr]);
}

__global__ __launch_bounds__(256) void transpose_bf16_kernel(
    const bf16s* __restrict__ in, long long ldin, long long strideIn,
    bf16s* __restrict__ out, long long ldout, long long strideOut) {
  __shared__ bf16s tile[32][34];
  const int bz = blockIdx.z;
  in  += (size_t)bz * strideIn;
  out += (size_t)bz * strideOut;
  const int c0 = blockIdx.x * 32, r0 = blockIdx.y * 32;
  const int tx = threadIdx.x & 31, ty = threadIdx.x >> 5;
  for (int rr = ty; rr < 32; rr += 8)
    tile[rr][tx] = in[(size_t)(r0 + rr) * ldin + c0 + tx];
  __syncthreads();
  for (int rr = ty; rr < 32; rr += 8)
    out[(size_t)(c0 + rr) * ldout + r0 + tx] = tile[tx][rr];
}

__global__ __launch_bounds__(256) void tanh_gate_kernel(const float* __restrict__ ctx,
                                                        const float* __restrict__ gate,
                                                        bf16s* __restrict__ out,
                                                        size_t n, int dk) {
  for (size_t i = (size_t)blockIdx.x * blockDim.x + threadIdx.x; i < n;
       i += (size_t)gridDim.x * blockDim.x)
    out[i] = f2b(tanhf(ctx[i] * gate[i & (size_t)(dk - 1)]));
}

__global__ __launch_bounds__(256) void softmax_rows_kernel(const float* __restrict__ X,
                                                           bf16s* __restrict__ Y, int ncols) {
  const int row = blockIdx.x;
  const float* x = X + (size_t)row * ncols;
  bf16s* y = Y + (size_t)row * ncols;
  __shared__ float red[256];
  const int tid = threadIdx.x;
  float m = -3.4e38f;
  for (int c = tid; c < ncols; c += 256) m = fmaxf(m, x[c]);
  red[tid] = m;
  __syncthreads();
  for (int s = 128; s > 0; s >>= 1) {
    if (tid < s) red[tid] = fmaxf(red[tid], red[tid + s]);
    __syncthreads();
  }
  m = red[0];
  __syncthreads();
  float sum = 0.0f;
  for (int c = tid; c < ncols; c += 256) sum += __expf(x[c] - m);
  red[tid] = sum;
  __syncthreads();
  for (int s = 128; s > 0; s >>= 1) {
    if (tid < s) red[tid] += red[tid + s];
    __syncthreads();
  }
  const float inv = 1.0f / red[0];
  for (int c = tid; c < ncols; c += 256) y[c] = f2b(__expf(x[c] - m) * inv);
}

// final = softmax(scores + lw * cam_bias), lw = sigmoid(w0 + 0.5*w1)
__global__ __launch_bounds__(256) void combine_softmax_kernel(
    const float* __restrict__ scores, const float* __restrict__ cam_bias,
    const float* __restrict__ w0, const float* __restrict__ w1,
    bf16s* __restrict__ Y, int ncols) {
  const int row = blockIdx.x;
  const float* xs = scores + (size_t)row * ncols;
  const float* xb = cam_bias + (size_t)row * ncols;
  bf16s* y = Y + (size_t)row * ncols;
  const float lw = 1.0f / (1.0f + __expf(-(w0[0] + 0.5f * w1[0])));
  __shared__ float red[256];
  const int tid = threadIdx.x;
  float m = -3.4e38f;
  for (int c = tid; c < ncols; c += 256) m = fmaxf(m, xs[c] + lw * xb[c]);
  red[tid] = m;
  __syncthreads();
  for (int s = 128; s > 0; s >>= 1) {
    if (tid < s) red[tid] = fmaxf(red[tid], red[tid + s]);
    __syncthreads();
  }
  m = red[0];
  __syncthreads();
  float sum = 0.0f;
  for (int c = tid; c < ncols; c += 256) sum += __expf(xs[c] + lw * xb[c] - m);
  red[tid] = sum;
  __syncthreads();
  for (int s = 128; s > 0; s >>= 1) {
    if (tid < s) red[tid] += red[tid + s];
    __syncthreads();
  }
  const float inv = 1.0f / red[0];
  for (int c = tid; c < ncols; c += 256) y[c] = f2b(__expf(xs[c] + lw * xb[c] - m) * inv);
}

// ---------- orchestration ----------
extern "C" void kernel_launch(void* const* d_in, const int* in_sizes, int n_in,
                              void* d_out, int out_size, void* d_ws, size_t ws_size,
                              hipStream_t stream) {
  (void)in_sizes; (void)n_in; (void)out_size; (void)ws_size;
  const float* hs     = (const float*)d_in[0];
  const float* attn_w = (const float*)d_in[1];
  const float* attn_b = (const float*)d_in[2];
  const float* proj_w = (const float*)d_in[3];
  const float* proj_b = (const float*)d_in[4];
  const float* gate   = (const float*)d_in[5];
  const float* w0     = (const float*)d_in[6];
  const float* w1     = (const float*)d_in[7];
  float* outF = (float*)d_out;

  const int Bn = 2, S = 2048, E = 2048, E3 = 6144, DK = 128;
  const float SCALE = 0.0883883476483184f; // 1/sqrt(128)

  char* p = (char*)d_ws;
  auto take = [&](size_t bytes) -> char* {
    char* r = p;
    p += (bytes + 255) & ~(size_t)255;
    return r;
  };
  bf16s* hsB    = (bf16s*)take((size_t)Bn * S * E * 2);   // later reused as context
  bf16s* wqkvT  = (bf16s*)take((size_t)E3 * E * 2);       // c_attn_w^T  [6144,2048]
  bf16s* wprT   = (bf16s*)take((size_t)E * E * 2);        // c_proj_w^T  [2048,2048]
  bf16s* qkvB   = (bf16s*)take((size_t)Bn * S * E3 * 2);  // QKV bf16
  float* scores = (float*)take((size_t)Bn * S * S * 4);   // full Q.K^T * SCALE
  float* camS   = (float*)take((size_t)Bn * S * S * 4);   // cam scores, reused for cam_bias
  bf16s* baseA  = (bf16s*)take((size_t)Bn * S * S * 2);   // base_A, reused for final attn
  float* camCtx = (float*)take((size_t)Bn * S * DK * 4);
  bf16s* Tg     = (bf16s*)take((size_t)Bn * S * DK * 2);  // tanh(ctx*gate)
  bf16s* vcamT  = (bf16s*)take((size_t)Bn * DK * S * 2);  // V_cam^T [128,2048]
  bf16s* vT     = (bf16s*)take((size_t)Bn * E * S * 2);   // V^T [2048,2048]
  bf16s* ctxB   = hsB;                                    // reuse

  const long long sQKV = (long long)S * E3;
  const long long sSS  = (long long)S * S;

  // 1) hidden f32 -> bf16
  f32_to_bf16_kernel<<<4096, 256, 0, stream>>>(hs, hsB, (size_t)Bn * S * E);
  // 2) transpose weights (f32 -> bf16)
  transpose_f32_to_bf16_kernel<<<dim3(E3 / 32, E / 32, 1), 256, 0, stream>>>(attn_w, E3, wqkvT, E);
  transpose_f32_to_bf16_kernel<<<dim3(E / 32, E / 32, 1), 256, 0, stream>>>(proj_w, E, wprT, E);
  // 3) QKV = hs @ W_qkv + b   (M=4096, N=6144, K=2048) -> bf16
  wmma_gemm_nt<bf16s><<<dim3(E3 / BN, (Bn * S) / BM, 1), 256, 0, stream>>>(
      hsB, E, 0, wqkvT, E, 0, qkvB, E3, 0, Bn * S, E3, E, 1.0f, attn_b);
  // 4) scores = SCALE * Q @ K^T   (batched, K-matrix is rows at col offset E)
  wmma_gemm_nt<float><<<dim3(S / BN, S / BM, Bn), 256, 0, stream>>>(
      qkvB, E3, sQKV, qkvB + E, E3, sQKV, scores, S, sSS, S, S, E, SCALE, nullptr);
  // 5) cam scores = SCALE * Q_cam @ K_cam^T  (K = 128)
  wmma_gemm_nt<float><<<dim3(S / BN, S / BM, Bn), 256, 0, stream>>>(
      qkvB, E3, sQKV, qkvB + E, E3, sQKV, camS, S, sSS, S, S, DK, SCALE, nullptr);
  // 6) base_A = softmax(cam scores) -> bf16
  softmax_rows_kernel<<<Bn * S, 256, 0, stream>>>(camS, baseA, S);
  // 7) V_cam^T (bf16 transpose, batched)
  transpose_bf16_kernel<<<dim3(DK / 32, S / 32, Bn), 256, 0, stream>>>(
      qkvB + 2 * E, E3, sQKV, vcamT, S, (long long)DK * S);
  // 8) cam_ctx = base_A @ V_cam   (M=2048, N=128, K=2048) -> f32
  wmma_gemm_nt<float><<<dim3(DK / BN, S / BM, Bn), 256, 0, stream>>>(
      baseA, S, sSS, vcamT, S, (long long)DK * S, camCtx, DK, (long long)S * DK,
      S, DK, S, 1.0f, nullptr);
  // 9) Tg = tanh(cam_ctx * gate) -> bf16
  tanh_gate_kernel<<<2048, 256, 0, stream>>>(camCtx, gate, Tg, (size_t)Bn * S * DK, DK);
  // 10) cam_bias = SCALE * Tg @ K_cam^T  (K=128) -> reuse camS
  wmma_gemm_nt<float><<<dim3(S / BN, S / BM, Bn), 256, 0, stream>>>(
      Tg, DK, (long long)S * DK, qkvB + E, E3, sQKV, camS, S, sSS, S, S, DK, SCALE, nullptr);
  // 11) final_attn = softmax(scores + lw*cam_bias) -> bf16 (reuse baseA)
  combine_softmax_kernel<<<Bn * S, 256, 0, stream>>>(scores, camS, w0, w1, baseA, S);
  // 12) V^T (bf16 transpose, batched)
  transpose_bf16_kernel<<<dim3(E / 32, S / 32, Bn), 256, 0, stream>>>(
      qkvB + 2 * E, E3, sQKV, vT, S, (long long)S * E);
  // 13) context = final_attn @ V   (M=2048, N=2048, K=2048) -> bf16 (reuse hsB)
  wmma_gemm_nt<bf16s><<<dim3(E / BN, S / BM, Bn), 256, 0, stream>>>(
      baseA, S, sSS, vT, S, (long long)S * E, ctxB, E, (long long)S * E,
      S, E, S, 1.0f, nullptr);
  // 14) output = context @ c_proj_w + b  (M=4096, N=2048, K=2048) -> f32 d_out
  wmma_gemm_nt<float><<<dim3(E / BN, (Bn * S) / BM, 1), 256, 0, stream>>>(
      ctxB, E, 0, wprT, E, 0, outF, E, 0, Bn * S, E, E, 1.0f, proj_b);
}
